// SNN_GlucoseNetwork_54082228191948
// MI455X (gfx1250) — matrix operand order repaired and checked by reading, not compiled, not run on hardware
//
#include <hip/hip_runtime.h>
#include <hip/hip_bf16.h>

// ---------------------------------------------------------------------------
// SNN (3-layer LIF, snntorch 'zero' reset) forward over T=512 steps.
// Persistent-state WMMA kernel for gfx1250 (CDNA5, wave32).
//   B=1024, T=512, D=64, H1=256, H2=128, O=16
// One block = one 16-row batch tile (WMMA M), 8 waves, 64 blocks.
//   L1: wave w owns cols [32w,32w+32)  -> 2 Ntile x 2 Ktile WMMAs
//   L2: wave w owns cols [16w,16w+16)  -> 8 Ktile WMMAs (K=256)
//   L3: wave 0 only, 4 Ktile WMMAs (K=128)
// Weights converted to f16 once (pre-pass) and held in VGPRs for all 512
// steps. Membrane state held in f32 accumulator fragments (bias folded into
// the WMMA C operand). Spikes exchanged through LDS as f16 (exact: 0/1).
// Input tile is double-buffered: the global load for step t+1 is issued at
// the top of step t and committed to LDS at the end of step t, hiding the
// HBM/L2 latency behind a full step of matrix work. 2 barriers per step.
// ---------------------------------------------------------------------------

typedef __attribute__((ext_vector_type(16))) _Float16 v16h;
typedef __attribute__((ext_vector_type(8)))  float    v8f;

namespace {
constexpr int Bn  = 1024;
constexpr int Tn  = 512;
constexpr int Dn  = 64;
constexpr int H1n = 256;
constexpr int H2n = 128;
constexpr int On  = 16;
constexpr float THRESH = 1.0f;

constexpr int W1_ELTS = H1n * Dn;    // 16384
constexpr int W2_ELTS = H2n * H1n;   // 32768
constexpr int W3_ELTS = On  * H2n;   // 2048
}

union AFrag {
    v16h     v;
    unsigned u[8];
};

// Build a 16x32 f16 A-operand fragment from a row-major LDS row.
// Layout (16-bit A, wave32): lane = (lo,hi); VGPR j holds K-pair:
//   K = 2j + 16*hi + kbase  (contiguous, matches B fragments)
// Consecutive j are consecutive words -> vectorizes to b128.
__device__ __forceinline__ v16h make_afrag(const _Float16* row, int kbase, int hi) {
    AFrag a;
#pragma unroll
    for (int j = 0; j < 8; ++j) {
        const int k = kbase + 2 * j + 16 * hi;
        a.u[j] = *reinterpret_cast<const unsigned*>(row + k);
    }
    return a.v;
}

// 32x16 f16 B-operand fragment: lane (lo,hi) = column lo, K-half hi; the 16
// halfs per lane are contiguous in k for row-major [N,K] weights -> one
// 32-byte load per fragment.
__device__ __forceinline__ v16h load_bfrag(const _Float16* p) {
    return *reinterpret_cast<const v16h*>(p);
}

__device__ __forceinline__ v8f splat8(float v) {
    v8f r;
#pragma unroll
    for (int i = 0; i < 8; ++i) r[i] = v;
    return r;
}

// -------------------------- weight f32 -> f16 pre-pass ---------------------
__global__ void convert_weights_kernel(const float* __restrict__ W1,
                                       const float* __restrict__ W2,
                                       const float* __restrict__ W3,
                                       _Float16* __restrict__ dst) {
    const int i = blockIdx.x * blockDim.x + threadIdx.x;
    if (i < W1_ELTS) {
        dst[i] = (_Float16)W1[i];
    } else if (i < W1_ELTS + W2_ELTS) {
        dst[i] = (_Float16)W2[i - W1_ELTS];
    } else if (i < W1_ELTS + W2_ELTS + W3_ELTS) {
        dst[i] = (_Float16)W3[i - (W1_ELTS + W2_ELTS)];
    }
}

// ------------------------------ main persistent kernel ---------------------
__launch_bounds__(256, 1)
__global__ void snn_lif3_forward_kernel(const float* __restrict__ x,
                                        const _Float16* __restrict__ W1h,
                                        const _Float16* __restrict__ W2h,
                                        const _Float16* __restrict__ W3h,
                                        const float* __restrict__ b1,
                                        const float* __restrict__ b2,
                                        const float* __restrict__ b3,
                                        const float* __restrict__ pbeta1,
                                        const float* __restrict__ pbeta2,
                                        const float* __restrict__ pbeta3,
                                        float* __restrict__ out) {
    __shared__ _Float16 xh[2][16 * Dn];  // 4 KB : double-buffered f16 input tile
    __shared__ _Float16 s1[16 * H1n];    // 8 KB : layer-1 spikes
    __shared__ _Float16 s2[16 * H2n];    // 4 KB : layer-2 spikes

    const int tid  = threadIdx.x;
    const int wave = tid >> 5;
    const int lane = tid & 31;
    const int lo   = lane & 15;   // D/C col within tile; A row
    const int hi   = lane >> 4;   // K-half select
    const int bbase = blockIdx.x * 16;

    const float beta1 = fminf(fmaxf(pbeta1[0], 0.0f), 1.0f);
    const float beta2 = fminf(fmaxf(pbeta2[0], 0.0f), 1.0f);
    const float beta3 = fminf(fmaxf(pbeta3[0], 0.0f), 1.0f);

    // ---- preload all weight B-fragments into VGPRs (reused 512x) ----
    const int c1base = wave * 32;   // layer-1 column range of this wave
    v16h Bw1[2][2];
#pragma unroll
    for (int nt = 0; nt < 2; ++nt)
#pragma unroll
        for (int kt = 0; kt < 2; ++kt)
            Bw1[nt][kt] = load_bfrag(W1h + (c1base + 16 * nt + lo) * Dn + 32 * kt + 16 * hi);

    const int c2base = wave * 16;   // layer-2 column range of this wave
    v16h Bw2[8];
#pragma unroll
    for (int kt = 0; kt < 8; ++kt)
        Bw2[kt] = load_bfrag(W2h + (c2base + lo) * H1n + 32 * kt + 16 * hi);

    v16h Bw3[4];
    if (wave == 0) {
#pragma unroll
        for (int kt = 0; kt < 4; ++kt)
            Bw3[kt] = load_bfrag(W3h + lo * H2n + 32 * kt + 16 * hi);
    }

    // Bias folded into WMMA C-operand: per-lane splat (D/C layout broadcasts
    // one column per lane across the 8 row elements).
    const v8f bs1a = splat8(b1[c1base + lo]);
    const v8f bs1b = splat8(b1[c1base + 16 + lo]);
    const v8f bs2  = splat8(b2[c2base + lo]);
    const v8f bs3  = splat8((wave == 0) ? b3[lo] : 0.0f);

    // LIF membrane state lives in accumulator-fragment layout:
    // lane (lo,hi), vector elem r  ->  (row = r + 8*hi, col = lo)
    v8f m1a = {}; v8f m1b = {}; v8f m2 = {}; v8f m3 = {}; v8f sum3 = {};

    // x staging: 256 threads x float4 = the 16x64 f32 tile, coalesced.
    const int xrow = tid >> 4;
    const int xcol = (tid & 15) * 4;
    const float* xrp = x + ((size_t)(bbase + xrow) * Tn) * Dn + xcol;

    // ---- pre-stage tile t=0 ----
    {
        const float4 xv = *reinterpret_cast<const float4*>(xrp);
        union { _Float16 h[4]; unsigned long long q; } pk;
        pk.h[0] = (_Float16)xv.x; pk.h[1] = (_Float16)xv.y;
        pk.h[2] = (_Float16)xv.z; pk.h[3] = (_Float16)xv.w;
        *reinterpret_cast<unsigned long long*>(&xh[0][xrow * Dn + xcol]) = pk.q;
    }
    __builtin_prefetch(xrp + (size_t)Dn, 0, 1);  // warm t=1
    __syncthreads();

    for (int t = 0; t < Tn; ++t) {
        // ---- issue global load for step t+1 now; commit to LDS at step end.
        float4 xnext;
        const bool have_next = (t + 1 < Tn);
        if (have_next)
            xnext = *reinterpret_cast<const float4*>(xrp + (size_t)(t + 1) * Dn);
        if (t + 2 < Tn)
            __builtin_prefetch(xrp + (size_t)(t + 2) * Dn, 0, 1);  // global_prefetch_b8

        const _Float16* xcur = xh[t & 1];

        // ================= layer 1: [16x64] @ [64x256]^T =================
        {
            const v16h A0 = make_afrag(&xcur[lo * Dn], 0,  hi);
            const v16h A1 = make_afrag(&xcur[lo * Dn], 32, hi);
            v8f a0 = bs1a;  // C = bias
            v8f a1 = bs1b;
            a0 = __builtin_amdgcn_wmma_f32_16x16x32_f16(false, A0, false, Bw1[0][0], (short)0, a0, false, false);
            a0 = __builtin_amdgcn_wmma_f32_16x16x32_f16(false, A1, false, Bw1[0][1], (short)0, a0, false, false);
            a1 = __builtin_amdgcn_wmma_f32_16x16x32_f16(false, A0, false, Bw1[1][0], (short)0, a1, false, false);
            a1 = __builtin_amdgcn_wmma_f32_16x16x32_f16(false, A1, false, Bw1[1][1], (short)0, a1, false, false);
#pragma unroll
            for (int r = 0; r < 8; ++r) {
                // LIF zero-reset: reset computed from PREVIOUS membrane
                float mo   = m1a[r];
                float keep = (mo > THRESH) ? 0.0f : 1.0f;
                float mn   = (beta1 * mo + a0[r]) * keep;
                m1a[r] = mn;
                s1[(r + 8 * hi) * H1n + c1base + lo] = (mn > THRESH) ? (_Float16)1.0f : (_Float16)0.0f;

                mo   = m1b[r];
                keep = (mo > THRESH) ? 0.0f : 1.0f;
                mn   = (beta1 * mo + a1[r]) * keep;
                m1b[r] = mn;
                s1[(r + 8 * hi) * H1n + c1base + 16 + lo] = (mn > THRESH) ? (_Float16)1.0f : (_Float16)0.0f;
            }
        }
        __syncthreads();   // barrier A: s1 ready

        // ================= layer 2: [16x256] @ [256x128]^T ===============
        {
            v8f a2 = bs2;  // C = bias
#pragma unroll
            for (int kt = 0; kt < 8; ++kt) {
                const v16h A = make_afrag(&s1[lo * H1n], 32 * kt, hi);
                a2 = __builtin_amdgcn_wmma_f32_16x16x32_f16(false, A, false, Bw2[kt], (short)0, a2, false, false);
            }
#pragma unroll
            for (int r = 0; r < 8; ++r) {
                const float mo   = m2[r];
                const float keep = (mo > THRESH) ? 0.0f : 1.0f;
                const float mn   = (beta2 * mo + a2[r]) * keep;
                m2[r] = mn;
                s2[(r + 8 * hi) * H2n + c2base + lo] = (mn > THRESH) ? (_Float16)1.0f : (_Float16)0.0f;
            }
        }

        // ---- commit next input tile (other buffer; no reader until after
        //      barrier B, writes race with nothing) ----
        if (have_next) {
            union { _Float16 h[4]; unsigned long long q; } pk;
            pk.h[0] = (_Float16)xnext.x; pk.h[1] = (_Float16)xnext.y;
            pk.h[2] = (_Float16)xnext.z; pk.h[3] = (_Float16)xnext.w;
            *reinterpret_cast<unsigned long long*>(&xh[(t + 1) & 1][xrow * Dn + xcol]) = pk.q;
        }
        __syncthreads();   // barrier B: s2 + next x tile ready

        // ================= layer 3: [16x128] @ [128x16]^T (wave 0) =======
        if (wave == 0) {
            v8f a3 = bs3;  // C = bias
#pragma unroll
            for (int kt = 0; kt < 4; ++kt) {
                const v16h A = make_afrag(&s2[lo * H2n], 32 * kt, hi);
                a3 = __builtin_amdgcn_wmma_f32_16x16x32_f16(false, A, false, Bw3[kt], (short)0, a3, false, false);
            }
#pragma unroll
            for (int r = 0; r < 8; ++r) {
                const float mo   = m3[r];
                const float keep = (mo > THRESH) ? 0.0f : 1.0f;
                const float mn   = (beta3 * mo + a3[r]) * keep;
                m3[r] = mn;
                sum3[r] += mn;   // time-accumulate layer-3 membrane
            }
        }
        // wave 0 cannot pass barrier A(t+1) until its s2 reads are complete,
        // and s2 is only rewritten after barrier A(t+1): race-free.
    }

    if (wave == 0) {
        const float inv = 1.0f / (float)Tn;
#pragma unroll
        for (int r = 0; r < 8; ++r)
            out[(size_t)(bbase + r + 8 * hi) * On + lo] = sum3[r] * inv;
    }
}

// ---------------------------------------------------------------------------
extern "C" void kernel_launch(void* const* d_in, const int* in_sizes, int n_in,
                              void* d_out, int out_size, void* d_ws, size_t ws_size,
                              hipStream_t stream) {
    (void)in_sizes; (void)n_in; (void)out_size; (void)ws_size;

    const float* x   = (const float*)d_in[0];
    const float* W1  = (const float*)d_in[1];
    const float* b1  = (const float*)d_in[2];
    const float* W2  = (const float*)d_in[3];
    const float* b2  = (const float*)d_in[4];
    const float* W3  = (const float*)d_in[5];
    const float* b3  = (const float*)d_in[6];
    const float* be1 = (const float*)d_in[7];
    const float* be2 = (const float*)d_in[8];
    const float* be3 = (const float*)d_in[9];

    _Float16* wsh = (_Float16*)d_ws;            // 51200 halfs = 100 KB scratch
    _Float16* W1h = wsh;
    _Float16* W2h = wsh + W1_ELTS;
    _Float16* W3h = wsh + W1_ELTS + W2_ELTS;

    const int total = W1_ELTS + W2_ELTS + W3_ELTS;
    convert_weights_kernel<<<(total + 255) / 256, 256, 0, stream>>>(W1, W2, W3, wsh);

    snn_lif3_forward_kernel<<<Bn / 16, 256, 0, stream>>>(
        x, W1h, W2h, W3h, b1, b2, b3, be1, be2, be3, (float*)d_out);
}
